// cmdnet_bin_36936718745660
// MI455X (gfx1250) — matrix operand order repaired and checked by reading, not compiled, not run on hardware
//
#include <hip/hip_runtime.h>
#include <hip/hip_bf16.h>

// CDNA5 / gfx1250: per-wave32 batch. HH = H^T H and yH = y^T H both computed with
// v_wmma_f32_16x16x32_bf16 (fp32 accumulate); HH stays resident in WMMA accumulator
// VGPRs across all 10 iterations; per-iteration matvec done from the documented
// C/D register layout with LDS-broadcast xt and hardware TRANS math.

typedef __attribute__((ext_vector_type(16))) __bf16 v16bf;
typedef __attribute__((ext_vector_type(8)))  float  v8f;

#define NN 64
#define KK 32
#define ITER 10
#define WPB 8          // waves (batches) per 256-thread block

__device__ __forceinline__ float fast_tanh(float x) {
    // tanh(x) = (e^{2x}-1)/(e^{2x}+1); clamp so exp2 stays finite.
    x = fminf(fmaxf(x, -15.0f), 15.0f);
    float e = __builtin_amdgcn_exp2f(x * 2.8853900817779268f);  // exp(2x)
    return (e - 1.0f) * __builtin_amdgcn_rcpf(e + 1.0f);
}

__global__ __launch_bounds__(256) void cmdnet_bin_kernel(
    const float* __restrict__ yt,     // [B,64]
    const float* __restrict__ Ht,     // [B,64,32]
    const float* __restrict__ sig0,   // [B]
    const float* __restrict__ alphat, // [32]
    const float* __restrict__ taui,   // [11]
    const float* __restrict__ delta,  // [10]
    float* __restrict__ out,          // ft [B,32,2] then xt [B,32]
    int B)
{
    __shared__ float sH[WPB][NN * KK];   // H[n][k] per wave, f32
    __shared__ float sy[WPB][NN];
    __shared__ float sx[WPB][KK];        // xt broadcast buffer per wave

    const int lane = threadIdx.x & 31;
    const int wave = threadIdx.x >> 5;
    const int b    = blockIdx.x * WPB + wave;
    if (b >= B) return;                  // wave-uniform: EXEC stays all-1s below

    float* H = sH[wave];
    float* Y = sy[wave];
    float* X = sx[wave];

    // ---- stage H (8KB) and y (256B) into LDS, coalesced b128 ----
    const float* Hb = Ht + (size_t)b * (NN * KK);
    #pragma unroll
    for (int i = 0; i < 16; ++i) {
        float4 v = *(const float4*)(Hb + i * 128 + lane * 4);
        *(float4*)(H + i * 128 + lane * 4) = v;
    }
    const float* yb = yt + (size_t)b * NN;
    Y[lane]      = yb[lane];
    Y[lane + 32] = yb[lane + 32];
    // Same-wave LDS ops are pipeline in-order; just stop compiler reordering.
    asm volatile("" ::: "memory");

    const int  m  = lane & 15;
    const bool hi = lane >= 16;
    const int  hb = hi ? 8 : 0;

    // ---- build bf16 WMMA fragments from LDS ----
    // A[kt][nc]: 16x32 tile of H^T  (A[mm, kk] = H[nc*32+kk][kt*16+mm])
    v16bf Afr[2][2];
    #pragma unroll
    for (int kt = 0; kt < 2; ++kt) {
        const int k = kt * 16 + m;
        #pragma unroll
        for (int nc = 0; nc < 2; ++nc) {
            #pragma unroll
            for (int j = 0; j < 8; ++j) {
                int kb = (j < 4) ? ((hi ? 8 : 0) + 2 * j)
                                 : ((hi ? 24 : 16) + 2 * (j - 4));
                int n = nc * 32 + kb;
                Afr[kt][nc][2 * j]     = (__bf16)H[n * KK + k];
                Afr[kt][nc][2 * j + 1] = (__bf16)H[(n + 1) * KK + k];
            }
        }
    }
    // Bfr[nc][lt]: 32x16 tile of H  (B[kk, nn] = H[nc*32+kk][lt*16+nn])
    v16bf Bfr[2][2];
    #pragma unroll
    for (int nc = 0; nc < 2; ++nc) {
        #pragma unroll
        for (int lt = 0; lt < 2; ++lt) {
            const int l = lt * 16 + m;
            #pragma unroll
            for (int j = 0; j < 8; ++j) {
                int n = nc * 32 + ((j < 4) ? ((hi ? 8 : 0) + 2 * j)
                                           : ((hi ? 24 : 16) + 2 * (j - 4)));
                Bfr[nc][lt][2 * j]     = (__bf16)H[n * KK + l];
                Bfr[nc][lt][2 * j + 1] = (__bf16)H[(n + 1) * KK + l];
            }
        }
    }
    // Ay[nc]: every row = y, so every accumulator row of y-WMMA equals yH.
    v16bf Ay[2];
    #pragma unroll
    for (int nc = 0; nc < 2; ++nc) {
        #pragma unroll
        for (int j = 0; j < 8; ++j) {
            int kb = (j < 4) ? ((hi ? 8 : 0) + 2 * j)
                             : ((hi ? 24 : 16) + 2 * (j - 4));
            int n = nc * 32 + kb;
            Ay[nc][2 * j]     = (__bf16)Y[n];
            Ay[nc][2 * j + 1] = (__bf16)Y[n + 1];
        }
    }

    // ---- HH = H^T H : 2x2 tiles of 16x16, K=64 in two K=32 WMMA steps ----
    v8f acc[2][2];
    #pragma unroll
    for (int kt = 0; kt < 2; ++kt) {
        #pragma unroll
        for (int lt = 0; lt < 2; ++lt) {
            v8f c = {};
            c = __builtin_amdgcn_wmma_f32_16x16x32_bf16(
                    false, Afr[kt][0], false, Bfr[0][lt], (short)0, c, false, false);
            c = __builtin_amdgcn_wmma_f32_16x16x32_bf16(
                    false, Afr[kt][1], false, Bfr[1][lt], (short)0, c, false, false);
            acc[kt][lt] = c;
        }
    }
    // ---- yH = y^T H via WMMA (reuses B fragments); own-lane extraction ----
    float yH;
    {
        v8f c0 = {}, c1 = {};
        c0 = __builtin_amdgcn_wmma_f32_16x16x32_bf16(
                 false, Ay[0], false, Bfr[0][0], (short)0, c0, false, false);
        c0 = __builtin_amdgcn_wmma_f32_16x16x32_bf16(
                 false, Ay[1], false, Bfr[1][0], (short)0, c0, false, false);
        c1 = __builtin_amdgcn_wmma_f32_16x16x32_bf16(
                 false, Ay[0], false, Bfr[0][1], (short)0, c1, false, false);
        c1 = __builtin_amdgcn_wmma_f32_16x16x32_bf16(
                 false, Ay[1], false, Bfr[1][1], (short)0, c1, false, false);
        // lane k<16: yH[k] = col k of tile lt=0 (N=lane); lane k>=16: col k-16 of lt=1.
        yH = hi ? c1[0] : c0[0];
    }

    // ---- iteration state (lane k holds stream k) ----
    const float s0    = sig0[b];
    const float sig2  = s0 * s0;
    const float ra    = __builtin_amdgcn_rcpf(alphat[lane]);
    const float logit = __builtin_amdgcn_logf(ra - 1.0f) * 0.6931471805599453f;

    float s     = 0.0f;
    float t_abs = fabsf(taui[0]);
    float xt    = fast_tanh((logit + s) * 0.5f * t_abs);

    #pragma unroll
    for (int it = 0; it < ITER; ++it) {
        const float d  = delta[it];
        const float tn = fabsf(taui[it + 1]);

        // broadcast xt through LDS: 1 store + 4 b128 loads per lane
        asm volatile("" ::: "memory");
        X[lane] = xt;
        asm volatile("" ::: "memory");

        // xHH[l] = sum_k xt[k]*HH[k,l] from WMMA C/D layout:
        // this lane holds cols c0 = lane&15 (lt=0) and c1 = 16+(lane&15) (lt=1),
        // rows M = r + hb in acc[kt][lt][r].
        float p0 = 0.0f, p1 = 0.0f;
        #pragma unroll
        for (int kt = 0; kt < 2; ++kt) {
            float4 xa = *(const float4*)(X + kt * 16 + hb);
            float4 xb = *(const float4*)(X + kt * 16 + hb + 4);
            const float xk[8] = {xa.x, xa.y, xa.z, xa.w, xb.x, xb.y, xb.z, xb.w};
            #pragma unroll
            for (int r = 0; r < 8; ++r) {
                p0 = fmaf(xk[r], acc[kt][0][r], p0);
                p1 = fmaf(xk[r], acc[kt][1][r], p1);
            }
        }
        p0 += __shfl_xor(p0, 16, 32);   // combine row-halves across lane pairs
        p1 += __shfl_xor(p1, 16, 32);
        const float xhh = hi ? p1 : p0; // lane k: k<16 -> column k (lt=0), else lt=1

        const float gx = 0.5f * t_abs * (1.0f - xt * xt);
        const float gL = sig2 * fast_tanh(s * 0.5f) + gx * (xhh - yH);
        s    -= d * gL;
        t_abs = tn;
        xt    = fast_tanh((logit + s) * 0.5f * t_abs);
    }

    // ---- outputs: ft [B,32,2] then xt [B,32], concatenated flat ----
    float* ft = out;
    float* xo = out + (size_t)B * KK * 2;
    ft[(size_t)b * (KK * 2) + lane * 2 + 0] = 0.5f * (1.0f - xt);
    ft[(size_t)b * (KK * 2) + lane * 2 + 1] = 0.5f * (1.0f + xt);
    xo[(size_t)b * KK + lane]               = xt;
}

extern "C" void kernel_launch(void* const* d_in, const int* in_sizes, int n_in,
                              void* d_out, int out_size, void* d_ws, size_t ws_size,
                              hipStream_t stream) {
    const float* yt     = (const float*)d_in[0];
    const float* Ht     = (const float*)d_in[1];
    const float* sig0   = (const float*)d_in[2];
    const float* alphat = (const float*)d_in[3];
    const float* taui   = (const float*)d_in[4];
    const float* delta  = (const float*)d_in[5];
    const int B = in_sizes[2];          // sigmat0 has one element per batch

    const int blocks = (B + WPB - 1) / WPB;
    cmdnet_bin_kernel<<<blocks, 256, 0, stream>>>(
        yt, Ht, sig0, alphat, taui, delta, (float*)d_out, B);
}
